// AMLNeuralNetwork3D_72361609003133
// MI455X (gfx1250) — matrix operand order repaired and checked by reading, not compiled, not run on hardware
//
#include <hip/hip_runtime.h>

// ---------------------------------------------------------------------------
// AMLNeuralNetwork3D forward for MI455X (gfx1250, wave32, WMMA)
//   h  = relu(sum_l x[l,b,g]*Wl[g,l] + bl[g])          (elementwise, -> bf16)
//   h  = relu(h @ W1^T + b1)   (bf16 WMMA, f32 accum)
//   h  = relu(h @ W2^T + b2)
//   out= relu(h @ W3^T + b3)   (f32 out)
// Data path: double-buffered TDM tensor_load_to_lds staging (DMA overlapped
// with compute via TENSORcnt), f32->bf16 cvt in LDS, ds_load_b128 fragments,
// v_wmma_f32_16x16x32_bf16 with f32 accumulators.
// Grid maps m-tiles to the fast block dimension so the 8 blocks sharing the
// same weight rows run back-to-back -> W streamed from HBM ~once per layer
// (192 MB L2 captures the reuse).
// ---------------------------------------------------------------------------

typedef __attribute__((ext_vector_type(16))) __bf16 v16bf;
typedef __attribute__((ext_vector_type(8)))  __bf16 v8bf;
typedef __attribute__((ext_vector_type(8)))  float  v8f;
typedef __attribute__((ext_vector_type(4)))  unsigned int v4u;
typedef __attribute__((ext_vector_type(8)))  unsigned int v8u;

#define GENES 8192u
#define BATCH 1024u

// low 32 bits of a generic pointer to LDS == LDS byte offset (aperture form)
__device__ __forceinline__ unsigned lds_byte_off(const void* p) {
  return (unsigned)(unsigned long long)p;
}

// Issue one 2-D TDM load: tile (tile1 rows x tile0 elems) of a row-major
// tensor (td1 x td0, row stride stride0 elems) into LDS at ldsAddr.
// D# layout per CDNA5 ISA ch.8 (group0 128b, group1 256b; 2-group form).
__device__ __forceinline__ void tdm_load_2d(unsigned ldsAddr, const void* gptr,
                                            unsigned dszCode,        // 1=2B, 2=4B
                                            unsigned td0, unsigned td1,
                                            unsigned tile0, unsigned tile1,
                                            unsigned long long stride0) {
  unsigned long long ga = (unsigned long long)gptr;
  v4u g0;
  g0.x = 1u;                                   // count=1 (valid user descriptor)
  g0.y = ldsAddr;                              // lds_addr [63:32]
  g0.z = (unsigned)ga;                         // global_addr lo
  g0.w = (unsigned)(ga >> 32) | (2u << 30);    // global_addr hi + type=2 (image)

  unsigned long long q0 = ((unsigned long long)dszCode << 16)            // data_size
                        | ((unsigned long long)(td0 & 0xFFFFu) << 48);   // tensor_dim0 lo
  unsigned long long q1 = ((unsigned long long)(td0 >> 16) & 0xFFFFull)  // tensor_dim0 hi
                        | ((unsigned long long)td1 << 16)                // tensor_dim1
                        | ((unsigned long long)(tile0 & 0xFFFFu) << 48); // tile_dim0
  unsigned long long q2 = ((unsigned long long)(tile1 & 0xFFFFu))        // tile_dim1
                        | ((stride0 & 0xFFFFFFFFull) << 32);             // dim0_stride lo
  unsigned long long q3 = (stride0 >> 32) & 0xFFFFull;                   // dim0_stride hi

  v8u g1;
  g1.s0 = (unsigned)q0; g1.s1 = (unsigned)(q0 >> 32);
  g1.s2 = (unsigned)q1; g1.s3 = (unsigned)(q1 >> 32);
  g1.s4 = (unsigned)q2; g1.s5 = (unsigned)(q2 >> 32);
  g1.s6 = (unsigned)q3; g1.s7 = (unsigned)(q3 >> 32);

  asm volatile("tensor_load_to_lds %0, %1" :: "s"(g0), "s"(g1) : "memory");
}

__device__ __forceinline__ unsigned short bf16_bits(float f) {
  __bf16 h = (__bf16)f;
  unsigned short u;
  __builtin_memcpy(&u, &h, 2);
  return u;
}

// --------------------------- Kernel 1: local linear -------------------------
__global__ __launch_bounds__(256) void local_relu_kernel(
    const float* __restrict__ x,        // [4][1024][8192]
    const float* __restrict__ Wl,       // [8192][4]
    const float* __restrict__ bl,       // [8192]
    __bf16* __restrict__ h0)            // [1024][8192] bf16
{
  const unsigned long long idx =
      (unsigned long long)blockIdx.x * 256ull + threadIdx.x;   // b*G + g
  const unsigned g = (unsigned)(idx & (GENES - 1u));
  const float4 w = ((const float4*)Wl)[g];
  const unsigned long long BG = (unsigned long long)BATCH * GENES;
  float acc = bl[g];
  acc += x[idx          ] * w.x;
  acc += x[idx +     BG ] * w.y;
  acc += x[idx + 2ull*BG] * w.z;
  acc += x[idx + 3ull*BG] * w.w;
  acc = fmaxf(acc, 0.0f);
  h0[idx] = (__bf16)acc;
}

// --------------------------- Kernels 2-4: GEMM ------------------------------
// out[m,n] = relu( sum_k A[m,k]*W[n,k] + bias[n] ),  M=1024, N=K=8192
// Block: 128x128 tile, BK=64, 256 threads = 8 waves in 2(M) x 4(N).
// Double-buffered TDM staging: stage i+1 issued before waiting for stage i.
template <bool OUT_BF16>
__global__ __launch_bounds__(256) void gemm_bias_relu_kernel(
    const __bf16* __restrict__ A,       // [1024][8192] bf16
    const float*  __restrict__ W,       // [8192][8192] f32 (row-major, W^T gemm)
    const float*  __restrict__ bias,    // [8192]
    void* __restrict__ Out)             // bf16 or f32 [1024][8192]
{
  __shared__ __attribute__((aligned(32))) __bf16 sA[2][128 * 64];  // 2x16 KB
  __shared__ __attribute__((aligned(32))) float  sW[2][128 * 64];  // 2x32 KB
  __shared__ __attribute__((aligned(32))) __bf16 sB[2][128 * 64];  // 2x16 KB

  const unsigned tid  = threadIdx.x;
  const unsigned lane = tid & 31u;
  const unsigned wave = tid >> 5;
  const unsigned wm   = wave & 1u;   // 64-row slab
  const unsigned wn   = wave >> 1;   // 32-col slab
  const unsigned hv   = lane >> 4;   // half-wave (K-half selector)
  const unsigned l16  = lane & 15u;

  const unsigned mBase = blockIdx.x * 128u;   // m-tile fastest: W-row sharers adjacent
  const unsigned nBase = blockIdx.y * 128u;
  const unsigned K = GENES;

  const v8f vzero = {0.f, 0.f, 0.f, 0.f, 0.f, 0.f, 0.f, 0.f};
  v8f acc[4][2];
#pragma unroll
  for (int i = 0; i < 4; ++i)
#pragma unroll
    for (int j = 0; j < 2; ++j) acc[i][j] = vzero;

  // ---- prologue: stage 0 into buffer 0 ----
  if (wave == 0) {
    tdm_load_2d(lds_byte_off(sA[0]), A + (unsigned long long)mBase * K,
                1u, K, BATCH, 64u, 128u, (unsigned long long)K);
    tdm_load_2d(lds_byte_off(sW[0]), W + (unsigned long long)nBase * K,
                2u, K, K, 64u, 128u, (unsigned long long)K);
  }

  unsigned buf = 0u;
  for (unsigned k0 = 0; k0 < K; k0 += 64u) {
    // ---- issue stage i+1 into the other buffer, then wait for stage i ----
    if (wave == 0) {
      const unsigned kn = (k0 + 64u < K) ? (k0 + 64u) : 0u;  // wrap: harmless
      tdm_load_2d(lds_byte_off(sA[buf ^ 1u]),
                  A + (unsigned long long)mBase * K + kn,
                  1u, K, BATCH, 64u, 128u, (unsigned long long)K);
      tdm_load_2d(lds_byte_off(sW[buf ^ 1u]),
                  W + (unsigned long long)nBase * K + kn,
                  2u, K, K, 64u, 128u, (unsigned long long)K);
      __builtin_amdgcn_s_wait_tensorcnt(2);   // in-order TDM: stage i complete
    }
    __syncthreads();

    // ---- convert the f32 weight tile to bf16 (packed pairs) in LDS ----
    {
      const float2* src = (const float2*)sW[buf];
      unsigned* dst = (unsigned*)sB[buf];
#pragma unroll
      for (unsigned i = 0; i < 16u; ++i) {
        const unsigned e = tid + i * 256u;    // 4096 float2 total
        float2 v = src[e];
        dst[e] = (unsigned)bf16_bits(v.x) | ((unsigned)bf16_bits(v.y) << 16);
      }
    }
    __syncthreads();

    // ---- 2 WMMA K-steps of 32 ----
#pragma unroll
    for (unsigned kk = 0; kk < 64u; kk += 32u) {
      union Frag { v16bf v; v8bf h[2]; };
      Frag fa[4], fb[2];
#pragma unroll
      for (int tm = 0; tm < 4; ++tm) {
        // A 16x32 bf16: lanes 0-15 -> K 0..7 & 16..23; lanes 16-31 -> K 8..15 & 24..31
        const __bf16* p = &sA[buf][(wm * 64u + tm * 16u + l16) * 64u + kk + hv * 8u];
        fa[tm].h[0] = *(const v8bf*)p;
        fa[tm].h[1] = *(const v8bf*)(p + 16);
      }
#pragma unroll
      for (int tn = 0; tn < 2; ++tn) {
        // B 32x16 bf16: lane = column N; lanes 0-15 -> K 0..7 & 16..23; lanes 16-31 -> K 8..15 & 24..31
        const __bf16* p = &sB[buf][(wn * 32u + tn * 16u + l16) * 64u + kk + hv * 8u];
        fb[tn].h[0] = *(const v8bf*)p;
        fb[tn].h[1] = *(const v8bf*)(p + 16);
      }
#pragma unroll
      for (int tm = 0; tm < 4; ++tm)
#pragma unroll
        for (int tn = 0; tn < 2; ++tn)
          acc[tm][tn] = __builtin_amdgcn_wmma_f32_16x16x32_bf16(
              false, fa[tm].v, false, fb[tn].v, (short)0, acc[tm][tn],
              false, false);
    }
    __syncthreads();   // compute done: buffer may be re-staged next iteration
    buf ^= 1u;
  }

  // ---- epilogue: bias + relu + store ----
#pragma unroll
  for (int tn = 0; tn < 2; ++tn) {
    const unsigned col = nBase + wn * 32u + tn * 16u + l16;
    const float bn = bias[col];
#pragma unroll
    for (int tm = 0; tm < 4; ++tm) {
#pragma unroll
      for (int r = 0; r < 8; ++r) {
        const unsigned row = mBase + wm * 64u + tm * 16u + (unsigned)r + hv * 8u;
        float v = fmaxf(acc[tm][tn][r] + bn, 0.0f);
        if (OUT_BF16)
          ((__bf16*)Out)[(unsigned long long)row * GENES + col] = (__bf16)v;
        else
          ((float*)Out)[(unsigned long long)row * GENES + col] = v;
      }
    }
  }
}

// ------------------------------- launcher -----------------------------------
extern "C" void kernel_launch(void* const* d_in, const int* in_sizes, int n_in,
                              void* d_out, int out_size, void* d_ws, size_t ws_size,
                              hipStream_t stream) {
  (void)in_sizes; (void)n_in; (void)out_size; (void)ws_size;
  const float* x  = (const float*)d_in[0];
  const float* Wl = (const float*)d_in[1];
  const float* bl = (const float*)d_in[2];
  const float* W1 = (const float*)d_in[3];
  const float* b1 = (const float*)d_in[4];
  const float* W2 = (const float*)d_in[5];
  const float* b2 = (const float*)d_in[6];
  const float* W3 = (const float*)d_in[7];
  const float* b3 = (const float*)d_in[8];
  float* out = (float*)d_out;

  const unsigned long long HB = (unsigned long long)BATCH * GENES;  // 8M elems
  __bf16* h0 = (__bf16*)d_ws;           // 16 MB
  __bf16* h1 = h0 + HB;                 // 16 MB

  // layer 0: local linear + relu -> bf16
  local_relu_kernel<<<(BATCH * GENES) / 256u, 256, 0, stream>>>(x, Wl, bl, h0);

  dim3 grid(BATCH / 128u, GENES / 128u);   // 8 x 64 blocks (m fastest)
  // layer 1: h0 -> h1 (bf16)
  gemm_bias_relu_kernel<true ><<<grid, 256, 0, stream>>>(h0, W1, b1, (void*)h1);
  // layer 2: h1 -> h0 (bf16, reuse slot)
  gemm_bias_relu_kernel<true ><<<grid, 256, 0, stream>>>(h1, W2, b2, (void*)h0);
  // layer 3: h0 -> out (f32)
  gemm_bias_relu_kernel<false><<<grid, 256, 0, stream>>>(h0, W3, b3, (void*)out);
}
